// GraphWavenetConvolution_51728586113697
// MI455X (gfx1250) — matrix operand order repaired and checked by reading, not compiled, not run on hardware
//
#include <hip/hip_runtime.h>
#include <hip/hip_bf16.h>

// ---------------------------------------------------------------------------
// Graph WaveNet convolution for gfx1250 (MI455X).
//   terms = X1a + 2*A1*X1a + X1b + 2*A2*X1b + softmax_col(relu(Z Z^T)) X0^T - X0
//   out[b] = terms_node_major[:, b*32:(b+1)*32] @ W
// Attention part is flash-style fused on v_wmma_f32_16x16x32_bf16.
// ---------------------------------------------------------------------------

#define NNODES 8192
#define FDIM   512      // batch(16) * d_in(32)
#define NEDGES 262144
#define DEMB   64
#define DIN    32
#define DOUT   64
#define NBATCH 16

typedef __attribute__((ext_vector_type(16))) __bf16 bf16x16;
typedef __attribute__((ext_vector_type(8)))  __bf16 bf16x8;
typedef __attribute__((ext_vector_type(4)))  __bf16 bf16x4;
typedef __attribute__((ext_vector_type(8)))  float  floatx8;

__device__ inline floatx8 zerov8() {
  floatx8 r;
  for (int g = 0; g < 8; ++g) r[g] = 0.0f;
  return r;
}

__device__ inline floatx8 wmma_bf16(bf16x16 a, bf16x16 b, floatx8 c) {
  // D = A(16x32 bf16) * B(32x16 bf16) + C(16x16 f32)
  return __builtin_amdgcn_wmma_f32_16x16x32_bf16(
      /*neg_a=*/false, a, /*neg_b=*/false, b,
      /*c_mod=*/(short)0, c, /*reuse_a=*/false, /*reuse_b=*/false);
}

// A-operand loader from a row-major bf16 row, K window [k0, k0+32).
// Contiguous layout: lane<16 gets K {0..15}, lane>=16 gets K {16..31}
// — two adjacent 16B chunks per lane (works for global and LDS pointers).
__device__ inline bf16x16 load_Abf(const __bf16* row, int k0, int lh) {
  const bf16x8 lo = *(const bf16x8*)(row + k0 + lh * 16);
  const bf16x8 hi = *(const bf16x8*)(row + k0 + lh * 16 + 8);
  bf16x16 r;
  for (int e = 0; e < 8; ++e) { r[e] = lo[e]; r[8 + e] = hi[e]; }
  return r;
}

// ---------------------------------------------------------------------------
// prep: float -> bf16 copy (vectorized x4)
__global__ __launch_bounds__(256) void k_cvt_bf16(const float* __restrict__ src,
                                                  __bf16* __restrict__ dst, int n4) {
  int i = blockIdx.x * 256 + threadIdx.x;
  if (i < n4) {
    float4 v = ((const float4*)src)[i];
    bf16x4 o;
    o[0] = (__bf16)v.x; o[1] = (__bf16)v.y; o[2] = (__bf16)v.z; o[3] = (__bf16)v.w;
    ((bf16x4*)dst)[i] = o;
  }
}

// prep: X0[512][8192] -> V0[8192][512] (f32, node-major) and ACC = -X0^T
__global__ __launch_bounds__(256) void k_transpose_init(const float* __restrict__ X0,
                                                        float* __restrict__ V0,
                                                        float* __restrict__ ACCp) {
  __shared__ float t[32][33];
  const int tx = threadIdx.x & 31, ty = threadIdx.x >> 5;
  const int f0 = blockIdx.x * 32;   // 16 tiles over 512
  const int j0 = blockIdx.y * 32;   // 256 tiles over 8192
  for (int k = 0; k < 4; ++k)
    t[ty + 8 * k][tx] = X0[(size_t)(f0 + ty + 8 * k) * NNODES + (j0 + tx)];
  __syncthreads();
  for (int k = 0; k < 4; ++k) {
    float v = t[tx][ty + 8 * k];
    size_t o = (size_t)(j0 + ty + 8 * k) * FDIM + (f0 + tx);
    V0[o] = v;
    ACCp[o] = -v;   // the "- X0" term
  }
}

__global__ __launch_bounds__(256) void k_zero4(float4* __restrict__ p, int n4) {
  int i = blockIdx.x * 256 + threadIdx.x;
  if (i < n4) p[i] = make_float4(0.f, 0.f, 0.f, 0.f);
}

__global__ __launch_bounds__(256) void k_add4(float4* __restrict__ a,
                                              const float4* __restrict__ b, int n4) {
  int i = blockIdx.x * 256 + threadIdx.x;
  if (i < n4) {
    float4 x = a[i], y = b[i];
    x.x += y.x; x.y += y.y; x.z += y.z; x.w += y.w;
    a[i] = x;
  }
}

// COO SpMM scatter: Out[r,:] += scale * w_e * V[c,:]   (one wave per edge)
__global__ __launch_bounds__(256) void k_spmm(const int* __restrict__ rows,
                                              const int* __restrict__ cols,
                                              const float* __restrict__ wts,
                                              const float* __restrict__ Vsrc,
                                              float* __restrict__ Outp, float scale) {
  const int e = (blockIdx.x * 256 + threadIdx.x) >> 5;
  const int L = threadIdx.x & 31;
  const int r = rows[e], c = cols[e];
  const float wt = wts[e] * scale;
  const float* vp = Vsrc + (size_t)c * FDIM;
  float* op = Outp + (size_t)r * FDIM;
  for (int it = 0; it < 4; ++it) {
    int f = it * 128 + L * 4;
    float4 v = *(const float4*)(vp + f);
    atomicAdd(op + f + 0, wt * v.x);
    atomicAdd(op + f + 1, wt * v.y);
    atomicAdd(op + f + 2, wt * v.z);
    atomicAdd(op + f + 3, wt * v.w);
  }
}

// ---------------------------------------------------------------------------
// Column softmax stats of relu(Z Z^T): per column j, online max & sum-exp
// over all i, computed 16 columns per wave with WMMA score tiles.
__global__ __launch_bounds__(128) void k_stats(const __bf16* __restrict__ Zb,
                                               float* __restrict__ Mcol,
                                               float* __restrict__ invden) {
  const int tid = threadIdx.x;
  const int w = tid >> 5, L = tid & 31, lm = L & 15, lh = L >> 4;
  const int j0 = (blockIdx.x * 4 + w) * 16;

  // B operand (columns of S = rows of Z), fixed for the whole i loop.
  const __bf16* zc = Zb + (size_t)(j0 + lm) * DEMB;
  const bf16x16 B0 = *(const bf16x16*)(zc + lh * 16);
  const bf16x16 B1 = *(const bf16x16*)(zc + 32 + lh * 16);

  float m = 0.0f, l = 0.0f;     // relu >= 0, so 0 is a valid running max
  for (int i0 = 0; i0 < NNODES; i0 += 16) {
    const __bf16* zr = Zb + (size_t)(i0 + lm) * DEMB;
    bf16x16 A0 = load_Abf(zr, 0, lh);
    bf16x16 A1 = load_Abf(zr, 32, lh);
    floatx8 s = zerov8();
    s = wmma_bf16(A0, B0, s);
    s = wmma_bf16(A1, B1, s);
    float r[8], mnew = m;
    for (int g = 0; g < 8; ++g) { r[g] = fmaxf(s[g], 0.0f); mnew = fmaxf(mnew, r[g]); }
    float add = 0.0f;
    for (int g = 0; g < 8; ++g) add += __expf(r[g] - mnew);
    l = l * __expf(m - mnew) + add;
    m = mnew;
  }
  // lanes L and L^16 track the same column (different M halves): merge.
  float mo = __shfl_xor(m, 16, 32);
  float lo = __shfl_xor(l, 16, 32);
  float ms = fmaxf(m, mo);
  float ls = l * __expf(m - ms) + lo * __expf(mo - ms);
  if (lh == 0) {
    Mcol[j0 + lm] = ms;
    invden[j0 + lm] = 1.0f / ls;
  }
}

// ---------------------------------------------------------------------------
// Fused attention: ACC[i,f] += sum_j exp(relu(z_i.z_j)-M[j])*invden[j] * X0[f,j]
// Block = 4 waves, one 16-row i tile, all 512 features; j streamed 128/iter.
// Wave w builds its 16x32 P tile, stores it ROW-MAJOR to LDS (strided b16
// stores); every wave then re-loads all 4 tiles as WMMA A operands with two
// contiguous ds_load_b128 per k-step, and consumes them against contiguous
// B tiles from the pre-transposed bf16 V0^T (= bf16 X0).
__global__ __launch_bounds__(128) void k_attn(const __bf16* __restrict__ Zb,
                                              const __bf16* __restrict__ V0T,
                                              const float* __restrict__ Mcol,
                                              const float* __restrict__ invden,
                                              float* __restrict__ ACCp) {
  __shared__ __bf16 ldsP[4][16][32];   // [k-step][i row][k col], row-major
  const int tid = threadIdx.x;
  const int w = tid >> 5, L = tid & 31, lm = L & 15, lh = L >> 4;
  const int i0 = blockIdx.x * 16;

  // Score A operand (Z rows of this i tile) — invariant over the j loop.
  const __bf16* zr = Zb + (size_t)(i0 + lm) * DEMB;
  const bf16x16 As0 = load_Abf(zr, 0, lh);
  const bf16x16 As1 = load_Abf(zr, 32, lh);

  floatx8 acc[8];
  for (int ft = 0; ft < 8; ++ft) acc[ft] = zerov8();

  for (int jb = 0; jb < NNODES; jb += 128) {
    const int jw = jb + 32 * w;
    __builtin_prefetch(Zb + (size_t)(((jw + 128) & (NNODES - 1)) + lm) * DEMB, 0, 1);

    for (int t = 0; t < 2; ++t) {                       // two 16-col S tiles
      const int jt = jw + 16 * t;
      const __bf16* zc = Zb + (size_t)(jt + lm) * DEMB;
      bf16x16 B0 = *(const bf16x16*)(zc + lh * 16);
      bf16x16 B1 = *(const bf16x16*)(zc + 32 + lh * 16);
      floatx8 s = zerov8();
      s = wmma_bf16(As0, B0, s);
      s = wmma_bf16(As1, B1, s);
      const int jc = jt + lm;                           // D layout: n = lane%16
      const float mj = Mcol[jc];
      const float idj = invden[jc];
      // D layout: lane holds rows i = g + 8*lh, column k = 16t + lm.
      for (int g = 0; g < 8; ++g)
        ldsP[w][g + 8 * lh][16 * t + lm] =
            (__bf16)(__expf(fmaxf(s[g], 0.0f) - mj) * idj);
    }
    __syncthreads();

    for (int kk = 0; kk < 4; ++kk) {                    // 128-wide K, 4 steps
      bf16x16 Ap = load_Abf(&ldsP[kk][lm][0], 0, lh);   // 2x ds_load_b128
      const int joff = jb + kk * 32 + lh * 16;
      for (int ft = 0; ft < 8; ++ft) {                  // 128 features per wave
        const int f = 128 * w + 16 * ft + lm;
        bf16x16 Bp = *(const bf16x16*)(V0T + (size_t)f * NNODES + joff);
        acc[ft] = wmma_bf16(Ap, Bp, acc[ft]);
      }
    }
    __syncthreads();
  }

  for (int ft = 0; ft < 8; ++ft) {
    const int f = 128 * w + 16 * ft + lm;
    for (int g = 0; g < 8; ++g) {
      const size_t o = (size_t)(i0 + g + 8 * lh) * FDIM + f;
      ACCp[o] += acc[ft][g];
    }
  }
}

// ---------------------------------------------------------------------------
// out[b,j,o] = sum_d ACC[j, b*32+d] * W[d,o]   — one WMMA (K=32) per 16x16 tile
__global__ __launch_bounds__(128) void k_out(const float* __restrict__ ACCp,
                                             const float* __restrict__ Wm,
                                             float* __restrict__ outp) {
  __shared__ __bf16 Wl[DIN * DOUT];
  const int tid = threadIdx.x;
  for (int idx = tid; idx < DIN * DOUT; idx += 128) Wl[idx] = (__bf16)Wm[idx];
  __syncthreads();
  const int w = tid >> 5, L = tid & 31, lm = L & 15, lh = L >> 4;
  const int b = blockIdx.x >> 9;
  const int j0 = (blockIdx.x & 511) * 16;

  const float* ar = ACCp + (size_t)(j0 + lm) * FDIM + b * DIN;
  bf16x16 A;
  {
    const int kb = lh * 16;
    for (int e = 0; e < 8; ++e) {
      A[e]     = (__bf16)ar[kb + e];        // K {0..7} / {16..23}
      A[8 + e] = (__bf16)ar[8 + kb + e];    // K {8..15} / {24..31}
    }
  }
  bf16x16 B;                                // column o of W, K half by lane group
  const int o = w * 16 + lm;
  for (int e = 0; e < 16; ++e) B[e] = Wl[(lh * 16 + e) * DOUT + o];

  floatx8 d = wmma_bf16(A, B, zerov8());
  for (int g = 0; g < 8; ++g)
    outp[((size_t)b * NNODES + (j0 + g + 8 * lh)) * DOUT + o] = d[g];
}

// ---------------------------------------------------------------------------
extern "C" void kernel_launch(void* const* d_in, const int* in_sizes, int n_in,
                              void* d_out, int out_size, void* d_ws, size_t ws_size,
                              hipStream_t stream) {
  (void)in_sizes; (void)n_in; (void)out_size; (void)ws_size;
  const int*   erow = (const int*)d_in[0];
  const int*   ecol = (const int*)d_in[1];
  const float* ew   = (const float*)d_in[2];
  const float* X    = (const float*)d_in[3];   // [16,32,8192] == X0[512][8192]
  const float* Z    = (const float*)d_in[4];   // [8192,64]
  const float* W    = (const float*)d_in[5];   // [32,64]
  float* out = (float*)d_out;

  char* ws = (char*)d_ws;
  float*  V0   = (float*)(ws);                            // 16 MB node-major X0^T
  float*  Yb   = (float*)(ws + ((size_t)16 << 20));       // 16 MB first-hop buffer
  float*  ACCp = (float*)(ws + ((size_t)32 << 20));       // 16 MB accumulator
  __bf16* V0T  = (__bf16*)(ws + ((size_t)48 << 20));      //  8 MB bf16(X0) = V0^T
  __bf16* Zb   = (__bf16*)(ws + ((size_t)56 << 20));      //  1 MB bf16 Z
  float*  Mc   = (float*)(ws + ((size_t)57 << 20));       // 32 KB col max
  float*  idn  = (float*)(ws + ((size_t)57 << 20) + 32768); // 32 KB 1/denom

  const int n4_feat = (NNODES * FDIM) / 4;                // 1048576
  k_cvt_bf16<<<n4_feat / 256, 256, 0, stream>>>(X, V0T, n4_feat);
  const int n4_z = (NNODES * DEMB) / 4;
  k_cvt_bf16<<<n4_z / 256, 256, 0, stream>>>(Z, Zb, n4_z);
  k_transpose_init<<<dim3(FDIM / 32, NNODES / 32), 256, 0, stream>>>(X, V0, ACCp);

  for (int k = 0; k < 2; ++k) {
    const int* rk = erow + (size_t)k * NEDGES;
    const int* ck = ecol + (size_t)k * NEDGES;
    const float* wk = ew + (size_t)k * NEDGES;
    k_zero4<<<n4_feat / 256, 256, 0, stream>>>((float4*)Yb, n4_feat);
    k_spmm<<<NEDGES * 32 / 256, 256, 0, stream>>>(rk, ck, wk, V0, Yb, 1.0f);  // X1
    k_add4<<<n4_feat / 256, 256, 0, stream>>>((float4*)ACCp, (const float4*)Yb, n4_feat);
    k_spmm<<<NEDGES * 32 / 256, 256, 0, stream>>>(rk, ck, wk, Yb, ACCp, 2.0f); // 2*A*X1
  }

  k_stats<<<NNODES / 64, 128, 0, stream>>>(Zb, Mc, idn);
  k_attn<<<NNODES / 16, 128, 0, stream>>>(Zb, V0T, Mc, idn, ACCp);
  k_out<<<NBATCH * (NNODES / 16), 128, 0, stream>>>(ACCp, W, out);
}
